// BaseCapNet_50345606643759
// MI455X (gfx1250) — matrix operand order repaired and checked by reading, not compiled, not run on hardware
//
#include <hip/hip_runtime.h>

#define B_   64
#define H_   384
#define T_   128
#define L_   512
#define MBK_ 8
#define HID_ 256

typedef __attribute__((ext_vector_type(16))) _Float16 v16h;
typedef __attribute__((ext_vector_type(8)))  _Float16 v8h;
typedef __attribute__((ext_vector_type(8)))  float    v8f;

// ---------------------------------------------------------------------------
// K1: stage A = item_emb[seq] as f16, row-major (B*L x HID). seq = [hist|tgt].
// ---------------------------------------------------------------------------
__global__ void k_gather_f16(const int* __restrict__ hist,
                             const int* __restrict__ tgt,
                             const float* __restrict__ item_emb,
                             _Float16* __restrict__ Af) {
    int row = blockIdx.x;              // 0 .. B*L-1
    int b   = row >> 9;                // /512
    int l   = row & 511;
    int id  = (l < H_) ? hist[b * H_ + l] : tgt[b * T_ + (l - H_)];
    int t   = threadIdx.x;             // 0..255
    Af[(size_t)row * HID_ + t] = (_Float16)item_emb[(size_t)id * HID_ + t];
}

// ---------------------------------------------------------------------------
// K2: Wt[n][k] = trans_W[k][n] for k<256 (top half), converted to f16.
// ---------------------------------------------------------------------------
__global__ void k_wt_f16(const float* __restrict__ W, _Float16* __restrict__ Wt) {
    int n = blockIdx.x;
    int k = threadIdx.x;
    Wt[(size_t)n * HID_ + k] = (_Float16)W[(size_t)k * HID_ + n];
}

// ---------------------------------------------------------------------------
// K3: ubase[b][n] = trans_b[n] + sum_k user_emb[uid[b]][k] * W[256+k][n]
// ---------------------------------------------------------------------------
__global__ void k_user_base(const int* __restrict__ user_id,
                            const float* __restrict__ user_emb,
                            const float* __restrict__ W,
                            const float* __restrict__ bias,
                            float* __restrict__ ubase) {
    __shared__ float s_ue[HID_];
    int b = blockIdx.x;
    int n = threadIdx.x;
    s_ue[n] = user_emb[(size_t)user_id[b] * HID_ + n];
    __syncthreads();
    float acc = bias[n];
    for (int k = 0; k < HID_; ++k)
        acc += s_ue[k] * W[(size_t)(HID_ + k) * HID_ + n];
    ubase[b * HID_ + n] = acc;
}

// ---------------------------------------------------------------------------
// K4: hist_hat = Af(f16) @ Wt^T + ubase  via v_wmma_f32_16x16x32_f16.
// One 16x16 tile per wave; each wave handles 2 N-tiles sharing the A fragment.
// Grid: (B*L)/16 blocks of 256 threads (8 waves covering N=256).
// ---------------------------------------------------------------------------
__global__ __launch_bounds__(256)
void k_gemm_wmma(const _Float16* __restrict__ Af,
                 const _Float16* __restrict__ Wt,
                 const float* __restrict__ ubase,
                 float* __restrict__ hh) {
    int mtile = blockIdx.x;                 // 16 M-rows per block
    int w     = threadIdx.x >> 5;           // wave 0..7
    int lane  = threadIdx.x & 31;
    int g     = lane >> 4;                  // half-wave group
    int nl    = lane & 15;
    int b     = mtile >> 5;                 // batch of this M-tile (512/16=32 tiles per b)

    int nc0 = (w * 2) * 16 + nl;
    int nc1 = (w * 2 + 1) * 16 + nl;

    float u0 = ubase[b * HID_ + nc0];
    float u1 = ubase[b * HID_ + nc1];
    v8f acc0, acc1;
#pragma unroll
    for (int r = 0; r < 8; ++r) { acc0[r] = u0; acc1[r] = u1; }

    int m = lane & 15;
    const _Float16* arow = Af + (size_t)(mtile * 16 + m) * HID_;
    const _Float16* b0p  = Wt + (size_t)nc0 * HID_;
    const _Float16* b1p  = Wt + (size_t)nc1 * HID_;

#pragma unroll
    for (int kb = 0; kb < 8; ++kb) {
        int k0 = kb * 32;
        // A 16x32 layout: lane 16g+m holds K = {16g..16g+7} in halves 0..7,
        //                 K = {16g+8..16g+15} in halves 8..15.
        v8h alo = *(const v8h*)(arow + k0 + 16 * g);
        v8h ahi = *(const v8h*)(arow + k0 + 16 * g + 8);
        union { v16h v; v8h h[2]; } ua;
        ua.h[0] = alo; ua.h[1] = ahi;
        // B 32x16 layout: lane 16g+n holds column n, K = 16g + 0..15 contiguous.
        v16h bf0 = *(const v16h*)(b0p + k0 + 16 * g);
        v16h bf1 = *(const v16h*)(b1p + k0 + 16 * g);
        acc0 = __builtin_amdgcn_wmma_f32_16x16x32_f16(false, ua.v, false, bf0,
                                                      (short)0, acc0, false, false);
        acc1 = __builtin_amdgcn_wmma_f32_16x16x32_f16(false, ua.v, false, bf1,
                                                      (short)0, acc1, false, false);
    }

    // C/D layout: VGPR r <-> M = 8g + r, N = nl
    int mrow = mtile * 16 + 8 * g;
#pragma unroll
    for (int r = 0; r < 8; ++r) {
        hh[(size_t)(mrow + r) * HID_ + nc0] = acc0[r];
        hh[(size_t)(mrow + r) * HID_ + nc1] = acc1[r];
    }
}

// ---------------------------------------------------------------------------
// K5: dynamic routing, one block per batch element, 512 threads (16 waves).
// a,b in LDS (8x512); caps,c in LDS (8x256). hh streamed from L2.
// ---------------------------------------------------------------------------
__global__ __launch_bounds__(512)
void k_routing(const float* __restrict__ caps_in,
               const float* __restrict__ hh_all,
               const int* __restrict__ projp,
               float* __restrict__ out) {
    __shared__ float s_caps[MBK_][HID_];
    __shared__ float s_a[MBK_][L_];
    __shared__ float s_b[MBK_][L_];
    __shared__ float s_c[MBK_][HID_];
    __shared__ float s_c2[MBK_][HID_];
    __shared__ float s_G[64], s_Gi[64], s_P[64];
    __shared__ float s_aug[8][16];
    __shared__ float s_f[8];
    __shared__ float s_red[512];

    int b = blockIdx.x;
    int t = threadIdx.x;                      // 0..511 (== l index)
    int proj = projp[0];
    const float* hh = hh_all + (size_t)b * L_ * HID_;

    // caps_cur = caps * cap_mask (mask is all-ones in this problem)
    for (int i = 0; i < 4; ++i) {
        int idx = t + i * 512;
        s_caps[idx >> 8][idx & 255] = caps_in[(size_t)b * MBK_ * HID_ + idx];
    }
    __syncthreads();

    // a0[k][l] = dot(hh[l], caps[k])
    {
        float acc[MBK_];
#pragma unroll
        for (int k = 0; k < MBK_; ++k) acc[k] = 0.f;
        const float* row = hh + (size_t)t * HID_;
        for (int h = 0; h < HID_; ++h) {
            float v = row[h];
#pragma unroll
            for (int k = 0; k < MBK_; ++k) acc[k] += v * s_caps[k][h];
        }
#pragma unroll
        for (int k = 0; k < MBK_; ++k) s_a[k][t] = acc[k];
    }
    __syncthreads();

    for (int it = 0; it < 3; ++it) {
        float eps = (it < 2) ? 0.004f : 0.02f;

        // b = softmax_k(a)   (cap_mask all true)
        {
            float mx = -1e30f;
#pragma unroll
            for (int k = 0; k < MBK_; ++k) mx = fmaxf(mx, s_a[k][t]);
            float e[MBK_], s = 0.f;
#pragma unroll
            for (int k = 0; k < MBK_; ++k) { e[k] = __expf(s_a[k][t] - mx); s += e[k]; }
            float inv = 1.f / s;
#pragma unroll
            for (int k = 0; k < MBK_; ++k) s_b[k][t] = e[k] * inv;
        }
        __syncthreads();

        // c[k][h] = sum_l b[k][l] * hh[l][h]
        for (int i = 0; i < 4; ++i) {
            int idx = t + i * 512, k = idx >> 8, h = idx & 255;
            float acc = 0.f;
            for (int l = 0; l < L_; ++l) acc += s_b[k][l] * hh[(size_t)l * HID_ + h];
            s_c[k][h] = acc;
        }
        __syncthreads();

        if (proj) {
            // G = c c^T (8x8)
            if (t < 64) {
                int i = t >> 3, j = t & 7;
                float acc = 0.f;
                for (int h = 0; h < HID_; ++h) acc += s_c[i][h] * s_c[j][h];
                s_G[t] = acc;
            }
            __syncthreads();
            // Gi = inv(G + eps I), Gauss-Jordan (G+epsI is SPD -> no pivoting)
            if (t == 0) {
                for (int i = 0; i < 8; ++i)
                    for (int j = 0; j < 8; ++j) {
                        s_aug[i][j]     = s_G[i * 8 + j] + ((i == j) ? eps : 0.f);
                        s_aug[i][8 + j] = (i == j) ? 1.f : 0.f;
                    }
                for (int p = 0; p < 8; ++p) {
                    float inv = 1.f / s_aug[p][p];
                    for (int c = 0; c < 16; ++c) s_aug[p][c] *= inv;
                    for (int r = 0; r < 8; ++r) {
                        if (r == p) continue;
                        float f = s_aug[r][p];
                        for (int c = 0; c < 16; ++c) s_aug[r][c] -= f * s_aug[p][c];
                    }
                }
                for (int i = 0; i < 8; ++i)
                    for (int j = 0; j < 8; ++j) s_Gi[i * 8 + j] = s_aug[i][8 + j];
            }
            __syncthreads();
            // P = G @ Gi
            if (t < 64) {
                int i = t >> 3, j = t & 7;
                float acc = 0.f;
#pragma unroll
                for (int q = 0; q < 8; ++q) acc += s_G[i * 8 + q] * s_Gi[q * 8 + j];
                s_P[t] = acc;
            }
            __syncthreads();
            // c2 = c - P @ c   (old mask == 1 everywhere)
            for (int i = 0; i < 4; ++i) {
                int idx = t + i * 512, k = idx >> 8, h = idx & 255;
                float pt = 0.f;
#pragma unroll
                for (int j = 0; j < 8; ++j) pt += s_P[k * 8 + j] * s_c[j][h];
                s_c2[k][h] = s_c[k][h] - pt;
            }
        } else {
            for (int i = 0; i < 4; ++i) {
                int idx = t + i * 512;
                s_c2[idx >> 8][idx & 255] = s_c[idx >> 8][idx & 255];
            }
        }
        __syncthreads();

        // squash: caps = n/(n^2+1) * c2
        if (t < 8) {
            float n2 = 0.f;
            for (int h = 0; h < HID_; ++h) { float v = s_c2[t][h]; n2 += v * v; }
            float n = sqrtf(n2);
            s_f[t] = n / (n2 + 1.f);
        }
        __syncthreads();
        for (int i = 0; i < 4; ++i) {
            int idx = t + i * 512, k = idx >> 8, h = idx & 255;
            s_caps[k][h] = s_c2[k][h] * s_f[k];
        }
        __syncthreads();

        // a += hh @ caps^T  (only for it < 2)
        if (it < 2) {
            float acc[MBK_];
#pragma unroll
            for (int k = 0; k < MBK_; ++k) acc[k] = 0.f;
            const float* row = hh + (size_t)t * HID_;
            for (int h = 0; h < HID_; ++h) {
                float v = row[h];
#pragma unroll
                for (int k = 0; k < MBK_; ++k) acc[k] += v * s_caps[k][h];
            }
#pragma unroll
            for (int k = 0; k < MBK_; ++k) s_a[k][t] += acc[k];
            __syncthreads();
        }
    }

    // caps_out (cap_mask all-ones)
    for (int i = 0; i < 4; ++i) {
        int idx = t + i * 512;
        out[(size_t)b * MBK_ * HID_ + idx] = s_caps[idx >> 8][idx & 255];
    }

    // create_score[l] = log(sum_k b) - mean_k log(b); count(score<1) > 5
    {
        float s = 0.f, ml = 0.f;
#pragma unroll
        for (int k = 0; k < MBK_; ++k) {
            float bv = s_b[k][t];
            s += bv;
            ml += __logf(bv);
        }
        float score = __logf(s) - ml * (1.f / 8.f);
        s_red[t] = (score < 1.0f) ? 1.f : 0.f;
    }
    __syncthreads();
    for (int stride = 256; stride > 0; stride >>= 1) {
        if (t < stride) s_red[t] += s_red[t + stride];
        __syncthreads();
    }
    if (t == 0) out[(size_t)B_ * MBK_ * HID_ + b] = (s_red[0] > 5.0f) ? 1.f : 0.f;
}

// ---------------------------------------------------------------------------
extern "C" void kernel_launch(void* const* d_in, const int* in_sizes, int n_in,
                              void* d_out, int out_size, void* d_ws, size_t ws_size,
                              hipStream_t stream) {
    (void)in_sizes; (void)n_in; (void)out_size; (void)ws_size;

    const int*   user_id  = (const int*)d_in[0];
    const int*   hist     = (const int*)d_in[1];
    const int*   tgt      = (const int*)d_in[2];
    const float* caps     = (const float*)d_in[7];
    const float* item_emb = (const float*)d_in[10];
    const float* user_emb = (const float*)d_in[11];
    const float* trans_W  = (const float*)d_in[12];
    const float* trans_b  = (const float*)d_in[13];
    const int*   proj     = (const int*)d_in[14];

    // workspace layout (all offsets 32B-aligned):
    //   Af   : f16  (B*L x HID)        16,777,216 B
    //   Wt   : f16  (HID x HID)           131,072 B
    //   ub   : f32  (B x HID)              65,536 B
    //   hh   : f32  (B*L x HID)        33,554,432 B   (~50.5 MB total)
    char* ws = (char*)d_ws;
    _Float16* Af = (_Float16*)ws;
    _Float16* Wt = (_Float16*)(ws + (size_t)16777216);
    float*    ub = (float*)(ws + (size_t)16777216 + 131072);
    float*    hh = (float*)(ws + (size_t)16777216 + 131072 + 65536);

    k_gather_f16<<<dim3(B_ * L_), dim3(HID_), 0, stream>>>(hist, tgt, item_emb, Af);
    k_wt_f16<<<dim3(HID_), dim3(HID_), 0, stream>>>(trans_W, Wt);
    k_user_base<<<dim3(B_), dim3(HID_), 0, stream>>>(user_id, user_emb, trans_W, trans_b, ub);
    k_gemm_wmma<<<dim3((B_ * L_) / 16), dim3(256), 0, stream>>>(Af, Wt, ub, hh);
    k_routing<<<dim3(B_), dim3(512), 0, stream>>>(caps, hh, proj, (float*)d_out);
}
